// SelfAttention_33303176413557
// MI455X (gfx1250) — matrix operand order repaired and checked by reading, not compiled, not run on hardware
//
#include <hip/hip_runtime.h>

// ---------------------------------------------------------------------------
// MI455X (gfx1250) self-attention forward.
//  B=2, S=2048, E=768, H=12, D=64.  bf16 WMMA (f32 accum) everywhere.
//  All GEMM B-operands and attention K/V tiles are staged in LDS with
//  gfx1250 async copies (GLOBAL_LOAD_ASYNC_TO_LDS_B128, ASYNCcnt), double-
//  buffered so the DMA pipe overlaps the WMMA pipe.
// ---------------------------------------------------------------------------

typedef __attribute__((ext_vector_type(16))) __bf16 v16bf;
typedef __attribute__((ext_vector_type(8)))  __bf16 v8bf;
typedef __attribute__((ext_vector_type(8)))  float  v8f;
typedef __attribute__((ext_vector_type(8)))  float  f32x8;

#define BATCH 2
#define SEQ   2048
#define EMBED 768
#define HEADS 12
#define HDIM  64

__device__ __forceinline__ v16bf cat8(v8bf lo, v8bf hi) {
    v16bf a;
#pragma unroll
    for (int j = 0; j < 8; ++j) { a[j] = lo[j]; a[j + 8] = hi[j]; }
    return a;
}

__device__ __forceinline__ v8bf cvt8(const float* p) {
    f32x8 v = *(const f32x8*)p;
    v8bf r;
#pragma unroll
    for (int j = 0; j < 8; ++j) r[j] = (__bf16)v[j];
    return r;
}

__device__ __forceinline__ v8f wmma_bf16(v16bf a, v16bf b, v8f c) {
    // v_wmma_f32_16x16x32_bf16  (neg_a, A, neg_b, B, c_mod, C, reuse_a, reuse_b)
    return __builtin_amdgcn_wmma_f32_16x16x32_bf16(false, a, false, b, (short)0, c,
                                                   false, false);
}

// per-lane 16B async copy global -> LDS (GLOBAL_LOAD_ASYNC_TO_LDS_B128, ASYNCcnt)
__device__ __forceinline__ void async_b128(const void* gptr, void* lptr) {
    unsigned lds = (unsigned)(size_t)lptr;                 // low 32b = LDS offset
    unsigned long long ga = (unsigned long long)(size_t)gptr;
    asm volatile("global_load_async_to_lds_b128 %0, %1, off"
                 :: "v"(lds), "v"(ga) : "memory");
}
__device__ __forceinline__ void wait_async0() {
    asm volatile("s_wait_asynccnt 0x0" ::: "memory");
}

// reductions confined to a 16-lane half of the wave32 (C-layout row lives in a half)
__device__ __forceinline__ float half_max(float v) {
#pragma unroll
    for (int m = 1; m < 16; m <<= 1) v = fmaxf(v, __shfl_xor(v, m, 32));
    return v;
}
__device__ __forceinline__ float half_sum(float v) {
#pragma unroll
    for (int m = 1; m < 16; m <<= 1) v += __shfl_xor(v, m, 32);
    return v;
}

// ---------------------------------------------------------------------------
// Kernel 1: W[k][n] (f32) -> Wt[n][k] (bf16)
// ---------------------------------------------------------------------------
__global__ __launch_bounds__(256) void pack_weights(const float* __restrict__ W,
                                                    __bf16* __restrict__ Wt) {
    int i = blockIdx.x * 256 + threadIdx.x;      // 0 .. 768*768-1
    int n = i / EMBED, k = i % EMBED;
    Wt[i] = (__bf16)W[(size_t)k * EMBED + n];
}

// ---------------------------------------------------------------------------
// Kernel 2: projection GEMM  Out = X @ W
//   Block = 4 waves sharing one 64-col weight supertile (staged in LDS,
//   double-buffered async); wave w owns m-tile (mblk*4+w), 16x64 output.
//   mode 0: Out[b][h][s][d]  (Q, K)      mode 1: Out[b][h][d][s]  (V transposed)
// ---------------------------------------------------------------------------
__global__ __launch_bounds__(128) void proj_gemm(const float* __restrict__ X,
                                                 const __bf16* __restrict__ Wt,
                                                 __bf16* __restrict__ Out, int mode) {
    __shared__ __align__(16) __bf16 wblk[2][64 * 32];   // 64 cols x 32 K (bf16) x2

    const int tid = threadIdx.x;
    const int lane = tid & 31;
    const int wave = tid >> 5;
    const int hl = lane >> 4, ln = lane & 15;
    const int mblk = blockIdx.x / 12, ntile = blockIdx.x % 12;
    const int mtile = mblk * 4 + wave;

    const __bf16* wrows = Wt + (size_t)(ntile * 64) * EMBED;
    auto stage = [&](int bufi, int kk) {
#pragma unroll
        for (int j = 0; j < 2; ++j) {
            int c = tid + 128 * j;                  // 0..255: 64 rows x 4 chunks
            int row = c >> 2, c8 = c & 3;
            async_b128(wrows + (size_t)row * EMBED + kk + c8 * 8,
                       &wblk[bufi][row * 32 + c8 * 8]);
        }
    };

    const float* xrow = X + (size_t)(mtile * 16 + ln) * EMBED;
    v8f acc0 = {}, acc1 = {}, acc2 = {}, acc3 = {};

    stage(0, 0);                                    // prologue fill
    for (int i = 0; i < EMBED / 32; ++i) {
        const int kk = i * 32, cur = i & 1;
        wait_async0();                              // my buf[cur] copies done
        __syncthreads();                            // everyone's done
        if (i + 1 < EMBED / 32) stage(cur ^ 1, kk + 32);  // overlap next DMA
        if (kk + 32 < EMBED) __builtin_prefetch(xrow + kk + 32, 0, 1);

        v16bf a = cat8(cvt8(xrow + kk + 16 * hl), cvt8(xrow + kk + 16 * hl + 8));
        const __bf16* wl = &wblk[cur][ln * 32 + 16 * hl];   // +512 per 16 rows
        v16bf b0 = cat8(*(const v8bf*)(wl),        *(const v8bf*)(wl + 8));
        v16bf b1 = cat8(*(const v8bf*)(wl + 512),  *(const v8bf*)(wl + 520));
        v16bf b2 = cat8(*(const v8bf*)(wl + 1024), *(const v8bf*)(wl + 1032));
        v16bf b3 = cat8(*(const v8bf*)(wl + 1536), *(const v8bf*)(wl + 1544));
        acc0 = wmma_bf16(a, b0, acc0);
        acc1 = wmma_bf16(a, b1, acc1);
        acc2 = wmma_bf16(a, b2, acc2);
        acc3 = wmma_bf16(a, b3, acc3);
    }

    v8f acc[4] = {acc0, acc1, acc2, acc3};
#pragma unroll
    for (int nd = 0; nd < 4; ++nd) {
#pragma unroll
        for (int r = 0; r < 8; ++r) {
            int m = mtile * 16 + 8 * hl + r;     // C layout: lanes16-31 hold M=8..15
            int n = ntile * 64 + nd * 16 + ln;
            int b = m >> 11, s = m & (SEQ - 1);
            int h = n >> 6, d = n & (HDIM - 1);
            size_t idx = (mode == 0)
                ? ((((size_t)b * HEADS + h) * SEQ + s) * HDIM + d)
                : ((((size_t)b * HEADS + h) * HDIM + d) * SEQ + s);
            Out[idx] = (__bf16)acc[nd][r];
        }
    }
}

// ---------------------------------------------------------------------------
// Kernel 3: flash attention. Block = 4 waves sharing one (b,h); K/V tiles
//   double-buffered in LDS via async copies; each wave owns 16 queries.
// ---------------------------------------------------------------------------
__global__ __launch_bounds__(128) void attn_kernel(const __bf16* __restrict__ Qh,
                                                   const __bf16* __restrict__ Kh,
                                                   const __bf16* __restrict__ Vt,
                                                   __bf16* __restrict__ Ctx) {
    __shared__ __align__(16) __bf16 kbuf[2][32 * HDIM];   // 32 keys x 64 dims
    __shared__ __align__(16) __bf16 vbuf[2][HDIM * 32];   // 64 dims x 32 keys
    __shared__ __align__(16) __bf16 plds[4][16 * 32];     // per-wave P staging

    const int tid = threadIdx.x;
    const int lane = tid & 31;
    const int wave = tid >> 5;
    const int hl = lane >> 4, ln = lane & 15;
    const int bh = blockIdx.x >> 5;                 // 24 (b,h) pairs, 32 blocks each
    const int qt = (blockIdx.x & 31) * 4 + wave;    // q-tile in sequence
    const int b = bh / HEADS, h = bh % HEADS;

    const __bf16* Qp = Qh + (size_t)bh * SEQ * HDIM;
    const __bf16* Kp = Kh + (size_t)bh * SEQ * HDIM;
    const __bf16* Vp = Vt + (size_t)bh * HDIM * SEQ;
    __bf16* pl = &plds[wave][0];

    const float escale = 0.03608439182435161f;      // 1/sqrt(768), per reference

    // cooperative async stage of one 32-key block of K and V into LDS buffer
    auto stage = [&](int bufi, int kb) {
#pragma unroll
        for (int j = 0; j < 2; ++j) {
            int c = tid + 128 * j;                  // 0..255 chunk ids
            int krow = c >> 3, kc8 = c & 7;         // K: 32 rows x 8 chunks
            async_b128(Kp + (size_t)(kb + krow) * HDIM + kc8 * 8,
                       &kbuf[bufi][krow * HDIM + kc8 * 8]);
            int vrow = c >> 2, vc8 = c & 3;         // V: 64 rows x 4 chunks
            async_b128(Vp + (size_t)vrow * SEQ + kb + vc8 * 8,
                       &vbuf[bufi][vrow * 32 + vc8 * 8]);
        }
    };

    // Q A-fragments, K-dim = head_dim = 64 -> two 16x32 fragments
    const __bf16* qrow = Qp + (size_t)(qt * 16 + ln) * HDIM;
    v16bf aq0 = cat8(*(const v8bf*)(qrow + 16 * hl),
                     *(const v8bf*)(qrow + 16 * hl + 8));
    v16bf aq1 = cat8(*(const v8bf*)(qrow + 32 + 16 * hl),
                     *(const v8bf*)(qrow + 40 + 16 * hl));

    v8f o0 = {}, o1 = {}, o2 = {}, o3 = {};
    float mrun[8], lrun[8];
#pragma unroll
    for (int r = 0; r < 8; ++r) { mrun[r] = -1e30f; lrun[r] = 0.0f; }

    stage(0, 0);                                    // prologue fill

    for (int i = 0; i < SEQ / 32; ++i) {
        const int cur = i & 1;
        wait_async0();                              // my buf[cur] copies done
        __syncthreads();                            // block-wide rendezvous
        // kick next block's DMA so it overlaps this block's WMMAs
        if (i + 1 < SEQ / 32) stage(cur ^ 1, (i + 1) * 32);

        const __bf16* kcur = &kbuf[cur][0];
        const __bf16* vcur = &vbuf[cur][0];

        // ---- S = Q K^T for two 16-key tiles (K from LDS) -----------------
        v8f c0 = {}, c1 = {};
        {
            const __bf16* kr0 = kcur + ln * HDIM;
            v16bf bk = cat8(*(const v8bf*)(kr0 + 16 * hl),
                            *(const v8bf*)(kr0 + 16 * hl + 8));
            c0 = wmma_bf16(aq0, bk, c0);
            bk = cat8(*(const v8bf*)(kr0 + 32 + 16 * hl),
                      *(const v8bf*)(kr0 + 40 + 16 * hl));
            c0 = wmma_bf16(aq1, bk, c0);

            const __bf16* kr1 = kcur + (16 + ln) * HDIM;
            bk = cat8(*(const v8bf*)(kr1 + 16 * hl),
                      *(const v8bf*)(kr1 + 16 * hl + 8));
            c1 = wmma_bf16(aq0, bk, c1);
            bk = cat8(*(const v8bf*)(kr1 + 32 + 16 * hl),
                      *(const v8bf*)(kr1 + 40 + 16 * hl));
            c1 = wmma_bf16(aq1, bk, c1);
        }

        // ---- online softmax (rows live in 16-lane halves) ----------------
        float corr[8];
#pragma unroll
        for (int r = 0; r < 8; ++r) {
            float s0 = c0[r] * escale, s1 = c1[r] * escale;
            float bm = half_max(fmaxf(s0, s1));
            float mn = fmaxf(mrun[r], bm);
            float p0 = __expf(s0 - mn), p1 = __expf(s1 - mn);
            float rs = half_sum(p0 + p1);
            corr[r] = __expf(mrun[r] - mn);
            lrun[r] = lrun[r] * corr[r] + rs;
            mrun[r] = mn;
            // C-layout -> LDS row-major P[16][32] (bf16)
            pl[(8 * hl + r) * 32 + ln]      = (__bf16)p0;
            pl[(8 * hl + r) * 32 + 16 + ln] = (__bf16)p1;
        }
#pragma unroll
        for (int r = 0; r < 8; ++r) {
            o0[r] *= corr[r]; o1[r] *= corr[r];
            o2[r] *= corr[r]; o3[r] *= corr[r];
        }

        // wave-private LDS round trip: CDNA5 split counter wait
        asm volatile("s_wait_dscnt 0x0" ::: "memory");

        // P as A operand (16x32)
        const __bf16* prow = pl + ln * 32;
        v16bf pa = cat8(*(const v8bf*)(prow + 16 * hl),
                        *(const v8bf*)(prow + 16 * hl + 8));

        // ---- O += P * V  (V from LDS, dim-major so B operand contiguous) -
        {
            const __bf16* vr = vcur + (0 * 16 + ln) * 32 + 16 * hl;
            o0 = wmma_bf16(pa, cat8(*(const v8bf*)vr, *(const v8bf*)(vr + 8)), o0);
            vr = vcur + (1 * 16 + ln) * 32 + 16 * hl;
            o1 = wmma_bf16(pa, cat8(*(const v8bf*)vr, *(const v8bf*)(vr + 8)), o1);
            vr = vcur + (2 * 16 + ln) * 32 + 16 * hl;
            o2 = wmma_bf16(pa, cat8(*(const v8bf*)vr, *(const v8bf*)(vr + 8)), o2);
            vr = vcur + (3 * 16 + ln) * 32 + 16 * hl;
            o3 = wmma_bf16(pa, cat8(*(const v8bf*)vr, *(const v8bf*)(vr + 8)), o3);
        }
    }

    // ---- normalize and write context [B*S][768] (bf16) -------------------
#pragma unroll
    for (int r = 0; r < 8; ++r) {
        float inv = 1.0f / lrun[r];
        int m = qt * 16 + 8 * hl + r;
        size_t row = ((size_t)b * SEQ + m) * EMBED + h * HDIM + ln;
        Ctx[row + 0]  = (__bf16)(o0[r] * inv);
        Ctx[row + 16] = (__bf16)(o1[r] * inv);
        Ctx[row + 32] = (__bf16)(o2[r] * inv);
        Ctx[row + 48] = (__bf16)(o3[r] * inv);
    }
}

// ---------------------------------------------------------------------------
// Kernel 4: Y = Ctx @ Wo + bo   (Ctx bf16 [4096x768], Y f32)
//   Same LDS-staged, async double-buffered structure as proj_gemm.
// ---------------------------------------------------------------------------
__global__ __launch_bounds__(128) void out_gemm(const __bf16* __restrict__ Cx,
                                                const __bf16* __restrict__ Wot,
                                                const float* __restrict__ bo,
                                                float* __restrict__ Y) {
    __shared__ __align__(16) __bf16 wblk[2][64 * 32];

    const int tid = threadIdx.x;
    const int lane = tid & 31;
    const int wave = tid >> 5;
    const int hl = lane >> 4, ln = lane & 15;
    const int mblk = blockIdx.x / 12, ntile = blockIdx.x % 12;
    const int mtile = mblk * 4 + wave;

    const __bf16* wrows = Wot + (size_t)(ntile * 64) * EMBED;
    auto stage = [&](int bufi, int kk) {
#pragma unroll
        for (int j = 0; j < 2; ++j) {
            int c = tid + 128 * j;
            int row = c >> 2, c8 = c & 3;
            async_b128(wrows + (size_t)row * EMBED + kk + c8 * 8,
                       &wblk[bufi][row * 32 + c8 * 8]);
        }
    };

    const __bf16* crow = Cx + (size_t)(mtile * 16 + ln) * EMBED;
    v8f acc0 = {}, acc1 = {}, acc2 = {}, acc3 = {};

    stage(0, 0);
    for (int i = 0; i < EMBED / 32; ++i) {
        const int kk = i * 32, cur = i & 1;
        wait_async0();
        __syncthreads();
        if (i + 1 < EMBED / 32) stage(cur ^ 1, kk + 32);
        if (kk + 32 < EMBED) __builtin_prefetch(crow + kk + 32, 0, 1);

        v16bf a = cat8(*(const v8bf*)(crow + kk + 16 * hl),
                       *(const v8bf*)(crow + kk + 16 * hl + 8));
        const __bf16* wl = &wblk[cur][ln * 32 + 16 * hl];
        v16bf b0 = cat8(*(const v8bf*)(wl),        *(const v8bf*)(wl + 8));
        v16bf b1 = cat8(*(const v8bf*)(wl + 512),  *(const v8bf*)(wl + 520));
        v16bf b2 = cat8(*(const v8bf*)(wl + 1024), *(const v8bf*)(wl + 1032));
        v16bf b3 = cat8(*(const v8bf*)(wl + 1536), *(const v8bf*)(wl + 1544));
        acc0 = wmma_bf16(a, b0, acc0);
        acc1 = wmma_bf16(a, b1, acc1);
        acc2 = wmma_bf16(a, b2, acc2);
        acc3 = wmma_bf16(a, b3, acc3);
    }

    v8f acc[4] = {acc0, acc1, acc2, acc3};
#pragma unroll
    for (int nd = 0; nd < 4; ++nd) {
        int n = ntile * 64 + nd * 16 + ln;
        float bias = bo[n];
#pragma unroll
        for (int r = 0; r < 8; ++r) {
            int m = mtile * 16 + 8 * hl + r;
            Y[(size_t)m * EMBED + n] = acc[nd][r] + bias;
        }
    }
}

// ---------------------------------------------------------------------------
// Host launcher
// ---------------------------------------------------------------------------
extern "C" void kernel_launch(void* const* d_in, const int* in_sizes, int n_in,
                              void* d_out, int out_size, void* d_ws, size_t ws_size,
                              hipStream_t stream) {
    const float* value = (const float*)d_in[0];
    const float* key   = (const float*)d_in[1];
    const float* query = (const float*)d_in[2];
    const float* Wv    = (const float*)d_in[3];
    const float* Wk    = (const float*)d_in[4];
    const float* Wq    = (const float*)d_in[5];
    const float* Wo    = (const float*)d_in[6];
    const float* bo    = (const float*)d_in[7];
    float* Y = (float*)d_out;

    char* ws = (char*)d_ws;
    const size_t WBYTES = (size_t)EMBED * EMBED * 2;                 // 1,179,648
    const size_t HBYTES = (size_t)BATCH * HEADS * SEQ * HDIM * 2;    // 6,291,456
    __bf16* Wvt = (__bf16*)(ws);                ws += WBYTES;
    __bf16* Wkt = (__bf16*)(ws);                ws += WBYTES;
    __bf16* Wqt = (__bf16*)(ws);                ws += WBYTES;
    __bf16* Wot = (__bf16*)(ws);                ws += WBYTES;
    __bf16* Qh  = (__bf16*)(ws);                ws += HBYTES;
    __bf16* Kh  = (__bf16*)(ws);                ws += HBYTES;
    __bf16* Vt  = (__bf16*)(ws);                ws += HBYTES;
    __bf16* Ctx = (__bf16*)(ws);                ws += HBYTES;

    const int tblocks = (EMBED * EMBED) / 256;  // 2304
    pack_weights<<<tblocks, 256, 0, stream>>>(Wv, Wvt);
    pack_weights<<<tblocks, 256, 0, stream>>>(Wk, Wkt);
    pack_weights<<<tblocks, 256, 0, stream>>>(Wq, Wqt);
    pack_weights<<<tblocks, 256, 0, stream>>>(Wo, Wot);

    // 64 m-blocks x 12 n-tiles = 768 blocks, 4 waves each
    proj_gemm<<<768, 128, 0, stream>>>(query, Wqt, Qh, 0);
    proj_gemm<<<768, 128, 0, stream>>>(key,   Wkt, Kh, 0);
    proj_gemm<<<768, 128, 0, stream>>>(value, Wvt, Vt, 1);

    attn_kernel<<<768, 128, 0, stream>>>(Qh, Kh, Vt, Ctx);

    out_gemm<<<768, 128, 0, stream>>>(Ctx, Wot, bo, Y);
}
